// BinaryLinear_38147899523752
// MI455X (gfx1250) — matrix operand order repaired and checked by reading, not compiled, not run on hardware
//
#include <hip/hip_runtime.h>
#include <stdint.h>

#define IN_F  4096
#define OUT_F 4096
#define NROW  16384

#define BM 128
#define BN 256
#define BK 32
#define KSTEPS (IN_F / BK)          // 128
#define LSTRIDE 40                  // halfs per LDS tile row (32 + 8 pad) -> 80B
#define XTILE_BYTES (BM * LSTRIDE * 2)          // 10240 B
#define BUF_BYTES   ((BM + BN) * LSTRIDE * 2)   // 30720 B per double-buffer slot
#define BUF_HALFS   ((BM + BN) * LSTRIDE)       // 15360 halfs

typedef __attribute__((ext_vector_type(16))) _Float16 v16h;
typedef __attribute__((ext_vector_type(8)))  _Float16 v8h;
typedef __attribute__((ext_vector_type(8)))  float    v8f;
typedef __attribute__((ext_vector_type(4)))  float    v4f;

// ---------------- prep kernel 1: alpha + sign(W) in f16 ----------------
__global__ __launch_bounds__(256)
void binarize_weights(const float* __restrict__ W,
                      _Float16* __restrict__ wh,
                      float* __restrict__ alpha) {
    const int o = blockIdx.x;            // output row 0..4095
    const int t = threadIdx.x;           // 0..255
    const float* row = W + (size_t)o * IN_F;
    _Float16* orow = wh + (size_t)o * IN_F;
    float s = 0.f;
#pragma unroll
    for (int j = 0; j < IN_F / 256; ++j) {
        float w = row[t + j * 256];
        s += fabsf(w);
        // jnp.sign semantics: sign(0) == 0
        orow[t + j * 256] = (_Float16)((w > 0.f) ? 1.f : ((w < 0.f) ? -1.f : 0.f));
    }
    __shared__ float red[256];
    red[t] = s;
    __syncthreads();
#pragma unroll
    for (int off = 128; off > 0; off >>= 1) {
        if (t < off) red[t] += red[t + off];
        __syncthreads();
    }
    if (t == 0) alpha[o] = red[0] * (1.f / (float)IN_F);
}

// ---------------- prep kernel 2: x f32 -> f16 ----------------
__global__ __launch_bounds__(256)
void convert_x(const float* __restrict__ x, _Float16* __restrict__ xh) {
    size_t i = ((size_t)blockIdx.x * 256 + threadIdx.x) * 8;
    v4f a = *(const v4f*)(x + i);
    v4f b = *(const v4f*)(x + i + 4);
    v8h h;
#pragma unroll
    for (int j = 0; j < 4; ++j) {
        h[j]     = (_Float16)a[j];
        h[4 + j] = (_Float16)b[j];
    }
    *(v8h*)(xh + i) = h;
}

// ---------------- main GEMM: WMMA f16, async global->LDS double buffer ----
__global__ __launch_bounds__(256)
void binlin_gemm(const _Float16* __restrict__ xh,
                 const _Float16* __restrict__ wh,
                 const float* __restrict__ alpha,
                 const float* __restrict__ bias,
                 float* __restrict__ out) {
    // [buf][ x tile: BM rows | w tile: BN rows ], each row LSTRIDE halfs
    __shared__ _Float16 lds[2][BUF_HALFS];   // 60 KB

    const int tid   = threadIdx.x;       // 0..255
    const int lane  = tid & 31;
    const int wid   = tid >> 5;          // 0..7
    const int waveM = wid & 1;           // 2 row-groups of 64
    const int waveN = wid >> 1;          // 4 col-groups of 64
    const int l15   = lane & 15;
    const int hi    = lane >> 4;         // lane half selects K sub-range

    const int blockN = blockIdx.x;       // 0..15
    const int blockM = blockIdx.y;       // 0..127

    const _Float16* xbase = xh + (size_t)blockM * BM * IN_F;
    const _Float16* wbase = wh + (size_t)blockN * BN * IN_F;

    // async-copy chunk mapping (16B chunks, 4 per row of 32 halfs):
    //   x tile: 512 chunks -> 2 per thread ; w tile: 1024 chunks -> 4 per thread
    unsigned gxo[2], lxo[2], gwo[4], lwo[4];
#pragma unroll
    for (int t = 0; t < 2; ++t) {
        unsigned c   = (unsigned)tid + 256u * t;
        unsigned row = c >> 2, col = c & 3;
        gxo[t] = row * (IN_F * 2) + col * 16;
        lxo[t] = row * (LSTRIDE * 2) + col * 16;
    }
#pragma unroll
    for (int u = 0; u < 4; ++u) {
        unsigned c   = (unsigned)tid + 256u * u;
        unsigned row = c >> 2, col = c & 3;
        gwo[u] = row * (IN_F * 2) + col * 16;
        lwo[u] = XTILE_BYTES + row * (LSTRIDE * 2) + col * 16;
    }

    const unsigned ldsBase = (unsigned)(uintptr_t)(&lds[0][0]);

    v8f acc[4][4] = {};

    // issue one K-tile (both matrices) into buffer `buf`, then advance offsets
    auto issue = [&](int buf) {
        const unsigned bb = ldsBase + (unsigned)buf * BUF_BYTES;
        unsigned d;
#pragma unroll
        for (int t = 0; t < 2; ++t) {
            d = bb + lxo[t];
            asm volatile("global_load_async_to_lds_b128 %0, %1, %2"
                         :: "v"(d), "v"(gxo[t]), "s"(xbase) : "memory");
            gxo[t] += BK * 2;
        }
#pragma unroll
        for (int u = 0; u < 4; ++u) {
            d = bb + lwo[u];
            asm volatile("global_load_async_to_lds_b128 %0, %1, %2"
                         :: "v"(d), "v"(gwo[u]), "s"(wbase) : "memory");
            gwo[u] += BK * 2;
        }
    };

    issue(0);

    const int aoff = hi * 8;    // A: lanes 0-15 hold K 0-7/16-23, lanes 16-31 hold 8-15/24-31
    const int boff = hi * 8;    // B: lanes 0-15 hold K 0-7/16-23, lanes 16-31 hold 8-15/24-31

    for (int kt = 0; kt < KSTEPS; ++kt) {
        if (kt + 1 < KSTEPS) {
            issue((kt + 1) & 1);
            asm volatile("s_wait_asynccnt 0x6" ::: "memory"); // tile kt done; kt+1 in flight
        } else {
            asm volatile("s_wait_asynccnt 0x0" ::: "memory");
        }
        __syncthreads();

        const int buf = kt & 1;
        const _Float16* tX = &lds[buf][0];
        const _Float16* tW = &lds[buf][BM * LSTRIDE];

        v16h a[4];
#pragma unroll
        for (int m = 0; m < 4; ++m) {
            int r = waveM * 64 + m * 16 + l15;
            v8h alo = *(const v8h*)(tX + r * LSTRIDE + aoff);
            v8h ahi = *(const v8h*)(tX + r * LSTRIDE + 16 + aoff);
            a[m] = __builtin_shufflevector(alo, ahi,
                    0,1,2,3,4,5,6,7,8,9,10,11,12,13,14,15);
        }
#pragma unroll
        for (int n = 0; n < 4; ++n) {
            int r = waveN * 64 + n * 16 + l15;
            v8h blo = *(const v8h*)(tW + r * LSTRIDE + boff);
            v8h bhi = *(const v8h*)(tW + r * LSTRIDE + 16 + boff);
            v16h b = __builtin_shufflevector(blo, bhi,
                    0,1,2,3,4,5,6,7,8,9,10,11,12,13,14,15);
#pragma unroll
            for (int m = 0; m < 4; ++m)
                acc[m][n] = __builtin_amdgcn_wmma_f32_16x16x32_f16(
                    false, a[m], false, b, (short)0, acc[m][n], false, false);
        }

        __syncthreads();   // safe to overwrite this buffer next iteration
    }

    // epilogue: out[row][col] = acc * alpha[col] + bias[col]
#pragma unroll
    for (int n = 0; n < 4; ++n) {
        int col = blockN * BN + waveN * 64 + n * 16 + l15;
        float al = alpha[col];
        float bi = bias[col];
#pragma unroll
        for (int m = 0; m < 4; ++m) {
            int row0 = blockM * BM + waveM * 64 + m * 16 + hi * 8;
            float* op = out + (size_t)row0 * OUT_F + col;
#pragma unroll
            for (int v = 0; v < 8; ++v)
                op[(size_t)v * OUT_F] = acc[m][n][v] * al + bi;
        }
    }
}

// ---------------- host entry ----------------
extern "C" void kernel_launch(void* const* d_in, const int* in_sizes, int n_in,
                              void* d_out, int out_size, void* d_ws, size_t ws_size,
                              hipStream_t stream) {
    const float* x    = (const float*)d_in[0];   // [N, IN_F]
    const float* W    = (const float*)d_in[1];   // [OUT_F, IN_F]
    const float* bias = (const float*)d_in[2];   // [OUT_F]
    float* out = (float*)d_out;

    char* ws = (char*)d_ws;
    _Float16* xh   = (_Float16*)ws;                                   // 128 MB
    _Float16* wh   = (_Float16*)(ws + (size_t)NROW * IN_F * 2);       // +32 MB
    float*    alph = (float*)(ws + (size_t)NROW * IN_F * 2
                                 + (size_t)OUT_F * IN_F * 2);         // +16 KB

    convert_x<<<(NROW * (size_t)IN_F) / (256 * 8), 256, 0, stream>>>(x, xh);
    binarize_weights<<<OUT_F, 256, 0, stream>>>(W, wh, alph);
    binlin_gemm<<<dim3(OUT_F / BN, NROW / BM), 256, 0, stream>>>(xh, wh, alph, bias, out);
}
